// RVTDCNN_48988396978777
// MI455X (gfx1250) — compile-verified, matrix operand order checked by reading, not run on hardware
//
#include <hip/hip_runtime.h>
#include <hip/hip_bf16.h>
#include <math.h>

typedef __attribute__((ext_vector_type(16))) _Float16 v16h;
typedef __attribute__((ext_vector_type(8)))  _Float16 v8h;
typedef __attribute__((ext_vector_type(8)))  float    v8f;

#define B_      128
#define T_      4103   // sequence length of x
#define H_      6
#define OUTLEN_ 4096
#define FCIN_   192

// Branch-free tanh. Result feeds an f16 WMMA operand, so hardware-native
// precision is ample. Prefer CDNA5's V_TANH_F32 (single TRANS32 op).
__device__ __forceinline__ float fast_tanh(float v) {
#if __has_builtin(__builtin_amdgcn_tanhf)
    return __builtin_amdgcn_tanhf(v);
#else
    float a = fabsf(v);
    float e = __expf(2.0f * a);                       // v_mul + v_exp_f32
    float r = __builtin_amdgcn_rcpf(e + 1.0f);        // v_add + v_rcp_f32
    return copysignf(__builtin_fmaf(-2.0f, r, 1.0f), v);
#endif
}

// Per-lane K index for a v16h WMMA A/B fragment (16x32 f16 layout, ISA 7.12.2):
// lanes 0-15: elems 0..7 -> K0..7, elems 8..15 -> K16..23
// lanes 16-31: same +8
__device__ __forceinline__ int kmap(int e, int lane) {
    return (e & 7) + ((e >= 8) ? 16 : 0) + ((lane & 16) ? 8 : 0);
}

// Pre-pack constant B fragments into workspace (halves):
//  frag 0: conv B for output channels ch=0..15   (N = ch)
//  frag 1: conv B for output channels ch=16..31
//  frag 2..7: FC B for tap ow=0..5 (N = fc output, only n<2 nonzero)
// Layout: wsH[f*512 + lane*16 + e]
__global__ void rvtdcnn_pack_b(const float* __restrict__ conv_w,
                               const float* __restrict__ fc_w,
                               _Float16* __restrict__ wsH) {
    const int L = threadIdx.x;  // 0..31
    const int n = L & 15;
    #pragma unroll
    for (int half = 0; half < 2; ++half) {
        #pragma unroll
        for (int e = 0; e < 16; ++e) {
            int k = kmap(e, L);          // k = kw*6 + h, valid for k<18
            float v = 0.f;
            if (k < 18) {
                int kw = k / 6, h = k % 6;
                int ch = n + half * 16;  // ch = c*4 + oh
                int c = ch >> 2, oh = ch & 3;
                int kh = h - oh;
                if (kh >= 0 && kh < 3) v = conv_w[c * 9 + kh * 3 + kw];
            }
            wsH[half * 512 + L * 16 + e] = (_Float16)v;
        }
    }
    #pragma unroll
    for (int ow = 0; ow < 6; ++ow) {
        #pragma unroll
        for (int e = 0; e < 16; ++e) {
            int ch = kmap(e, L);         // K = channel index 0..31
            float v = (n < 2) ? fc_w[n * FCIN_ + ch * 6 + ow] : 0.f;
            wsH[(2 + ow) * 512 + L * 16 + e] = (_Float16)v;
        }
    }
}

// One wave per (batch, 16-timestep tile). 4 conv WMMAs + tanh + 6 FC WMMAs.
__global__ __launch_bounds__(32)
void rvtdcnn_main(const float* __restrict__ x,
                  const float* __restrict__ conv_b,
                  const float* __restrict__ fc_b,
                  const _Float16* __restrict__ wsH,
                  float* __restrict__ out) {
    __shared__ __align__(16) _Float16 smem[32 * 32]; // full2t rows t0..t0+31, 32 ch
    const int L  = threadIdx.x;
    const int b  = blockIdx.y;
    const int t0 = blockIdx.x * 16;
    const int m  = L & 15;                 // A-fragment row (timestep in tile)
    const int hi = (L & 16) ? 8 : 0;       // D-fragment row offset / K half select
    const int n  = L & 15;                 // B/C/D column

    // Constant B fragments (32B per lane, L2-resident)
    const v16h* bf = (const v16h*)wsH;
    v16h bc0 = bf[0 * 32 + L];
    v16h bc1 = bf[1 * 32 + L];

    // Build conv A fragments: A[m, k=(kw*6+h)] = x[b, t0+m+kw, h]
    v16h a0, a1;
    #pragma unroll
    for (int e = 0; e < 16; ++e) {
        int k = kmap(e, L);
        float v0 = 0.f, v1 = 0.f;
        if (k < 18) {
            int kw = k / 6, h = k % 6;
            int r0 = t0 + m + kw;          // <= 4097, in bounds
            int r1 = r0 + 16;              // may exceed; clamp (rows >t0+20 unused)
            if (r1 > T_ - 1) r1 = T_ - 1;
            v0 = x[(b * T_ + r0) * H_ + h];
            v1 = x[(b * T_ + r1) * H_ + h];
        }
        a0[e] = (_Float16)v0;
        a1[e] = (_Float16)v1;
    }

    v8f z = {};
    v8f d0 = __builtin_amdgcn_wmma_f32_16x16x32_f16(false, a0, false, bc0, (short)0, z, false, false);
    v8f d1 = __builtin_amdgcn_wmma_f32_16x16x32_f16(false, a0, false, bc1, (short)0, z, false, false);
    v8f d2 = __builtin_amdgcn_wmma_f32_16x16x32_f16(false, a1, false, bc0, (short)0, z, false, false);
    v8f d3 = __builtin_amdgcn_wmma_f32_16x16x32_f16(false, a1, false, bc1, (short)0, z, false, false);

    // bias + tanh, stage as f16 in LDS (row tl has 32 channels = 64B)
    const float bias0 = conv_b[n >> 2];
    const float bias1 = conv_b[(n + 16) >> 2];
    #pragma unroll
    for (int r = 0; r < 8; ++r) {
        int tl = r + hi;
        smem[tl * 32 + n]             = (_Float16)fast_tanh(d0[r] + bias0);
        smem[tl * 32 + n + 16]        = (_Float16)fast_tanh(d1[r] + bias1);
        smem[(tl + 16) * 32 + n]      = (_Float16)fast_tanh(d2[r] + bias0);
        smem[(tl + 16) * 32 + n + 16] = (_Float16)fast_tanh(d3[r] + bias1);
    }
    __syncthreads();

    // FC: out[t, n] = sum_ow sum_ch full2[t+ow, ch] * fc_w[n, ch*6+ow]
    v8f acc = {};
    const v8h* sp = (const v8h*)smem;
    const int bsel = (L & 16) ? 1 : 0;
    #pragma unroll
    for (int ow = 0; ow < 6; ++ow) {
        int row = m + ow;                          // <= 20 < 32
        v8h lo = sp[row * 4 + bsel];               // K = base..base+7
        v8h hv = sp[row * 4 + 2 + bsel];           // K = 16+base..16+base+7
        v16h a;
        #pragma unroll
        for (int i = 0; i < 8; ++i) { a[i] = lo[i]; a[i + 8] = hv[i]; }
        v16h bfc = bf[(2 + ow) * 32 + L];
        acc = __builtin_amdgcn_wmma_f32_16x16x32_f16(false, a, false, bfc, (short)0, acc, false, false);
    }

    // Store: only columns n<2 are real outputs (lanes 0,1,16,17)
    if (n < 2) {
        const float fb = fc_b[n];
        #pragma unroll
        for (int r = 0; r < 8; ++r) {
            int t = t0 + r + hi;
            out[(b * OUTLEN_ + t) * 2 + n] = acc[r] + fb;
        }
    }
}

extern "C" void kernel_launch(void* const* d_in, const int* in_sizes, int n_in,
                              void* d_out, int out_size, void* d_ws, size_t ws_size,
                              hipStream_t stream) {
    const float* x      = (const float*)d_in[0];   // [128, 4103, 6]
    const float* conv_w = (const float*)d_in[1];   // [8, 1, 3, 3]
    const float* conv_b = (const float*)d_in[2];   // [8]
    const float* fc_w   = (const float*)d_in[3];   // [2, 192]
    const float* fc_b   = (const float*)d_in[4];   // [2]
    float* out = (float*)d_out;                    // [128, 4096, 2]
    _Float16* wsH = (_Float16*)d_ws;               // 8 frags * 1KB = 8KB

    rvtdcnn_pack_b<<<1, 32, 0, stream>>>(conv_w, fc_w, wsH);
    rvtdcnn_main<<<dim3(OUTLEN_ / 16, B_), 32, 0, stream>>>(x, conv_b, fc_b, wsH, out);
}
